// SlidingAttentionModule_32435593019801
// MI455X (gfx1250) — compile-verified, matrix operand order checked by reading, not run on hardware
//
#include <hip/hip_runtime.h>

typedef __attribute__((ext_vector_type(16))) _Float16 v16h;
typedef __attribute__((ext_vector_type(8)))  float    v8f;
typedef __attribute__((ext_vector_type(4)))  int      v4i;

#define AS1 __attribute__((address_space(1)))
#define AS3 __attribute__((address_space(3)))

#if __has_builtin(__builtin_amdgcn_global_load_async_to_lds_b128)
#define SA_ASYNC_LDS 1
#else
#define SA_ASYNC_LDS 0
#endif

#if __has_builtin(__builtin_amdgcn_s_wait_asynccnt)
#define SA_WAIT_ASYNC() __builtin_amdgcn_s_wait_asynccnt(0)
#else
#define SA_WAIT_ASYNC() asm volatile("s_wait_asynccnt 0x0" ::: "memory")
#endif

// ---------------------------------------------------------------------------
// fp32 -> f16 conversion pass (x, w_qkv, w_out), vectorized float4 -> 4 halves.
// ~50 MB total traffic => ~2us at 23.3 TB/s; lets both GEMMs run a pure-f16
// async-copy pipeline afterwards.
// ---------------------------------------------------------------------------
__global__ __launch_bounds__(256)
void sliding_attn_cvt_f16(const float* __restrict__ src, _Float16* __restrict__ dst, int n4) {
    const int i = blockIdx.x * 256 + threadIdx.x;
    if (i < n4) {
        const float4 v = ((const float4*)src)[i];
        union { _Float16 h[4]; uint2 u; } cv;
        cv.h[0] = (_Float16)v.x; cv.h[1] = (_Float16)v.y;
        cv.h[2] = (_Float16)v.z; cv.h[3] = (_Float16)v.w;
        ((uint2*)dst)[i] = cv.u;
    }
}

// ---------------------------------------------------------------------------
// 16-byte global -> LDS copy. Async path uses GLOBAL_LOAD_ASYNC_TO_LDS_B128
// (tracked by ASYNCcnt); fallback is a plain register-staged uint4 copy.
// Builtin signature (from compiler diagnostics): (AS1 v4i*, AS3 v4i*, imm, imm)
// ---------------------------------------------------------------------------
__device__ __forceinline__ void copy_b128_to_lds(const _Float16* src, _Float16* dst) {
#if SA_ASYNC_LDS
    __builtin_amdgcn_global_load_async_to_lds_b128(
        (AS1 v4i*)(uintptr_t)(const void*)src,
        (AS3 v4i*)dst,
        /*imm offset=*/0, /*cpol=*/0);
#else
    *(uint4*)dst = *(const uint4*)src;
#endif
}

// ---------------------------------------------------------------------------
// Stage one 128x32-half tile (f16) into LDS: 512 chunks of 16B, 2 per thread.
// LDS row stride 40 halves (80B = 5*16B) keeps every chunk 16B-aligned.
// ---------------------------------------------------------------------------
__device__ __forceinline__
void stage_tile_f16(const _Float16* __restrict__ g, _Float16* l,
                    int row0, int K, int k0, int tid) {
#pragma unroll
    for (int i = 0; i < 2; ++i) {
        const int idx = tid + 256 * i;     // 0..511
        const int r   = idx >> 2;          // 0..127
        const int c   = (idx & 3) * 8;     // 0,8,16,24 halves
        copy_b128_to_lds(g + (size_t)(row0 + r) * K + k0 + c, l + r * 40 + c);
    }
}

// ---------------------------------------------------------------------------
// Fragment load from LDS per CDNA5 ISA 7.12.2 (16-bit A/B 16x32 layout):
//   lane l: row/col = l&15 ; kb = (l>=16)?8:0
//   halves 0..7  <- K = kb+0..7   ; halves 8..15 <- K = kb+16..kb+23
// Both ranges are contiguous 16B => two ds_load_b128 per fragment.
// ---------------------------------------------------------------------------
__device__ __forceinline__ v16h load_frag16(const _Float16* tile, int lane) {
    const int r  = lane & 15;
    const int kb = (lane >> 4) << 3;
    const _Float16* p = tile + r * 40 + kb;
    v16h f;
#pragma unroll
    for (int j = 0; j < 8; ++j) {
        f[j]     = p[j];
        f[j + 8] = p[j + 16];
    }
    return f;
}

// ---------------------------------------------------------------------------
// Tiled WMMA GEMM:  C[M,N] = A[M,K] * W[N,K]^T + bias[N]   (A, W f16)
// Block 128x128x32, 256 threads = 8 waves (4 in M x 2 in N), wave tile 32x64
// = 2x4 v_wmma_f32_16x16x32_f16 accumulators. Double-buffered LDS: async
// copies for slab k+1 are issued before the WMMAs of slab k, then
// s_wait_asynccnt 0 + barrier. Requires M%128==0, N%128==0, K%32==0.
// ---------------------------------------------------------------------------
template <bool OUT_F16>
__global__ __launch_bounds__(256)
void sliding_attn_gemm_wmma(const _Float16* __restrict__ A,
                            const _Float16* __restrict__ W,
                            const float* __restrict__ bias,
                            void* __restrict__ Cptr,
                            int M, int N, int K) {
    constexpr int BM = 128, BN = 128, BK = 32;
    constexpr int LDH = BK + 8;                          // 40 halves
    __shared__ __align__(16) _Float16 As[2][BM * LDH];   // 2 x 10 KB
    __shared__ __align__(16) _Float16 Bs[2][BN * LDH];   // 2 x 10 KB

    const int tid  = threadIdx.x;
    const int lane = tid & 31;
    const int wid  = tid >> 5;
    const int wm   = (wid & 3) * 32;   // wave row base within block tile
    const int wn   = (wid >> 2) * 64;  // wave col base within block tile
    const int row0 = blockIdx.y * BM;
    const int col0 = blockIdx.x * BN;

    v8f acc[2][4] = {};

    const int ksteps = K / BK;

    // prologue: stage slab 0 into buffer 0
    stage_tile_f16(A, &As[0][0], row0, K, 0, tid);
    stage_tile_f16(W, &Bs[0][0], col0, K, 0, tid);
#if SA_ASYNC_LDS
    SA_WAIT_ASYNC();
#endif
    __syncthreads();

    for (int ks = 0; ks < ksteps; ++ks) {
        const int b = ks & 1;

        // issue async copies for slab ks+1 into the other buffer; they
        // overlap the WMMA work below (previous barrier guarantees no wave
        // is still reading that buffer).
        if (ks + 1 < ksteps) {
            const int kn = (ks + 1) * BK;
            stage_tile_f16(A, &As[1 - b][0], row0, K, kn, tid);
            stage_tile_f16(W, &Bs[1 - b][0], col0, K, kn, tid);
        }

        // ---- WMMA compute on buffer b ----
        v16h af[2], bf[4];
#pragma unroll
        for (int tm = 0; tm < 2; ++tm)
            af[tm] = load_frag16(&As[b][(wm + tm * 16) * LDH], lane);
#pragma unroll
        for (int tn = 0; tn < 4; ++tn)
            bf[tn] = load_frag16(&Bs[b][(wn + tn * 16) * LDH], lane);

#pragma unroll
        for (int tm = 0; tm < 2; ++tm)
#pragma unroll
            for (int tn = 0; tn < 4; ++tn)
                acc[tm][tn] = __builtin_amdgcn_wmma_f32_16x16x32_f16(
                    /*neg_a=*/false, af[tm], /*neg_b=*/false, bf[tn],
                    /*c_mod=*/(short)0, acc[tm][tn],
                    /*reuse_a=*/false, /*reuse_b=*/false);

#if SA_ASYNC_LDS
        SA_WAIT_ASYNC();
#endif
        __syncthreads();
    }

    // ---- epilogue: C/D layout (VGPR r -> M = r + 8*(lane>=16), N = lane&15) ----
    const int colb = lane & 15;
    const int hi8  = (lane >> 4) * 8;
#pragma unroll
    for (int tm = 0; tm < 2; ++tm) {
#pragma unroll
        for (int tn = 0; tn < 4; ++tn) {
            const int col = col0 + wn + tn * 16 + colb;
            const float bv = bias[col];
            const int mbase = row0 + wm + tm * 16 + hi8;
#pragma unroll
            for (int r = 0; r < 8; ++r) {
                const float val = acc[tm][tn][r] + bv;
                if constexpr (OUT_F16)
                    ((_Float16*)Cptr)[(size_t)(mbase + r) * N + col] = (_Float16)val;
                else
                    ((float*)Cptr)[(size_t)(mbase + r) * N + col] = val;
            }
        }
    }
}

// ---------------------------------------------------------------------------
// Sliding-window attention (window = +/-8, <=17 keys/query): one wave32 per
// (b,h,q); lane owns 2 of the 64 head dims; shfl_xor tree reduces the dots.
// qkv: [B*S,3072] f16 (Q|K|V), ctx: [B*S,1024] f16.
// ---------------------------------------------------------------------------
__global__ __launch_bounds__(256)
void sliding_attn_window(const _Float16* __restrict__ qkv,
                         _Float16* __restrict__ ctx) {
    constexpr int S = 2048, LD = 3072, HD = 64, WIN = 17;
    const int gw   = (blockIdx.x * 256 + threadIdx.x) >> 5;  // 0..65535
    const int lane = threadIdx.x & 31;

    const int q = gw & (S - 1);
    const int h = (gw >> 11) & 15;
    const int b = gw >> 15;

    const size_t rowq = ((size_t)(b * S + q)) * LD + h * HD;
    const int d0 = lane * 2;

    const float qscale = 0.125f;  // 1/sqrt(64)
    const float q0 = (float)qkv[rowq + d0]     * qscale;
    const float q1 = (float)qkv[rowq + d0 + 1] * qscale;

    float s[WIN];
#pragma unroll
    for (int i = 0; i < WIN; ++i) {
        const int j = q - 8 + i;
        const bool valid = (j >= 0) && (j < S);
        const int jc = valid ? j : 0;
        const size_t kb = ((size_t)(b * S + jc)) * LD + 1024 + h * HD;
        float p = q0 * (float)qkv[kb + d0] + q1 * (float)qkv[kb + d0 + 1];
#pragma unroll
        for (int m = 16; m >= 1; m >>= 1) p += __shfl_xor(p, m, 32);
        s[i] = valid ? p : -1e30f;
    }

    float mx = -1e30f;
#pragma unroll
    for (int i = 0; i < WIN; ++i) mx = fmaxf(mx, s[i]);
    float sum = 0.f;
#pragma unroll
    for (int i = 0; i < WIN; ++i) { s[i] = __expf(s[i] - mx); sum += s[i]; }
    const float inv = 1.f / sum;

    float a0 = 0.f, a1 = 0.f;
#pragma unroll
    for (int i = 0; i < WIN; ++i) {
        const int j = q - 8 + i;
        const int jc = (j >= 0 && j < S) ? j : 0;
        const size_t vb = ((size_t)(b * S + jc)) * LD + 2048 + h * HD;
        const float w = s[i] * inv;
        a0 += w * (float)qkv[vb + d0];
        a1 += w * (float)qkv[vb + d0 + 1];
    }

    const size_t co = ((size_t)(b * S + q)) * 1024 + h * HD + d0;
    ctx[co]     = (_Float16)a0;
    ctx[co + 1] = (_Float16)a1;
}

// ---------------------------------------------------------------------------
// Launch: convert x/w_qkv/w_out to f16 -> QKV GEMM (WMMA, async-LDS) ->
// windowed attention -> out-proj GEMM (WMMA, async-LDS)
// ---------------------------------------------------------------------------
extern "C" void kernel_launch(void* const* d_in, const int* in_sizes, int n_in,
                              void* d_out, int out_size, void* d_ws, size_t ws_size,
                              hipStream_t stream) {
    (void)in_sizes; (void)n_in; (void)out_size; (void)ws_size;

    constexpr int Bb = 2, S = 2048, D = 1024;
    constexpr int M = Bb * S;          // 4096

    const float* x      = (const float*)d_in[0];
    // d_in[1] = token_ids (unused by the reference math)
    const float* w_qkv  = (const float*)d_in[2];
    const float* b_qkv  = (const float*)d_in[3];
    const float* w_out  = (const float*)d_in[4];
    const float* b_out  = (const float*)d_in[5];
    float*       out    = (float*)d_out;

    // workspace layout (f16): xh | wqkvh | wouth | qkv | ctx  (~48 MB)
    _Float16* xh    = (_Float16*)d_ws;                     // [4096,1024]
    _Float16* wqkvh = xh    + (size_t)M * D;               // [3072,1024]
    _Float16* wouth = wqkvh + (size_t)(3 * D) * D;         // [1024,1024]
    _Float16* qkv   = wouth + (size_t)D * D;               // [4096,3072]
    _Float16* ctx   = qkv   + (size_t)M * (3 * D);         // [4096,1024]

    dim3 blk(256);

    const int n4_x    = M * D / 4;
    const int n4_wqkv = 3 * D * D / 4;
    const int n4_wout = D * D / 4;
    sliding_attn_cvt_f16<<<(n4_x    + 255) / 256, blk, 0, stream>>>(x,     xh,    n4_x);
    sliding_attn_cvt_f16<<<(n4_wqkv + 255) / 256, blk, 0, stream>>>(w_qkv, wqkvh, n4_wqkv);
    sliding_attn_cvt_f16<<<(n4_wout + 255) / 256, blk, 0, stream>>>(w_out, wouth, n4_wout);

    // qkv = x @ w_qkv^T + b_qkv   (f16 in, f16 out)
    sliding_attn_gemm_wmma<true>
        <<<dim3((3 * D) / 128, M / 128), blk, 0, stream>>>(
            xh, wqkvh, b_qkv, qkv, M, 3 * D, D);

    // windowed softmax attention, 65536 waves, 8 waves/block
    sliding_attn_window<<<(Bb * 16 * S) / 8, blk, 0, stream>>>(qkv, ctx);

    // out = ctx @ w_out^T + b_out  (f16 in, fp32 out)
    sliding_attn_gemm_wmma<false>
        <<<dim3(D / 128, M / 128), blk, 0, stream>>>(
            ctx, wouth, b_out, out, M, D, D);
}
